// Adaptive_Scan_Mamba_55525337203210
// MI455X (gfx1250) — compile-verified, matrix operand order checked by reading, not run on hardware
//
#include <hip/hip_runtime.h>
#include <hip/hip_bf16.h>
#include <math.h>
#include <stdint.h>

// ---------------------------------------------------------------------------
// Adaptive Scan Mamba for MI455X (gfx1250, wave32, WMMA + TDM).
// GEMMs: A-tile staged by the Tensor Data Mover (tensor_load_to_lds, 6-arg
// clang-23 form), B-tile staged transposed as f16; v_wmma_f32_16x16x32_f16.
// Selective scans: register-resident recurrences with global prefetch.
// ---------------------------------------------------------------------------

typedef __attribute__((ext_vector_type(16))) _Float16 v16h;
typedef __attribute__((ext_vector_type(8)))  _Float16 v8h;
typedef __attribute__((ext_vector_type(8)))  float    v8f;
typedef __attribute__((ext_vector_type(4)))  unsigned int u32x4;
typedef __attribute__((ext_vector_type(8)))  int i32x8;
typedef __attribute__((ext_vector_type(4)))  int i32x4;

#if defined(__has_builtin)
#if __has_builtin(__builtin_amdgcn_tensor_load_to_lds)
#define GEMM_USE_TDM 1
#endif
#endif
#ifndef GEMM_USE_TDM
#define GEMM_USE_TDM 0
#endif

#define NSLICES 5

__device__ __forceinline__ int imin(int a, int b) { return a < b ? a : b; }

// ---------------- activation helpers ----------------
template<int ACT>
__device__ __forceinline__ float apply_act(float v) {
  if (ACT == 1) return fmaxf(v, 0.f);                          // relu
  if (ACT == 2) return (v > 20.f) ? v : log1pf(__expf(v));     // softplus
  if (ACT == 3) return 1.f / (1.f + __expf(-v));               // sigmoid
  return v;
}

__device__ __forceinline__ float silu_f(float v) {
  return v / (1.f + __expf(-v));
}

// ---------------- WMMA GEMM: C(M,N) = act(A(M,K) @ W(K,N) + bias) ----------
// A row-major (row stride lda); W row-major (K x N, stride N); C stride N.
// 32x32 C macro-tile per 128-thread block; 4 waves, one 16x16 WMMA tile each.
// M must be a multiple of 32 (true for all call sites: 4000, 12000).
template<int ACT>
__global__ void wmma_gemm_kernel(const float* __restrict__ A, int lda,
                                 const float* __restrict__ W,
                                 const float* __restrict__ bias,
                                 float* __restrict__ C,
                                 int M, int N, int K)
{
  __shared__ __align__(16) float    Asf[32][32];  // A tile [m][k], f32, TDM target
  __shared__ __align__(16) _Float16 Bst[32][40];  // B^T tile [n][k], f16

  const int tid    = threadIdx.x;
  const int waveId = tid >> 5;
  const int lane   = tid & 31;
  const int tilesN = (N + 31) >> 5;
  const int mt = blockIdx.x / tilesN;
  const int nt = blockIdx.x % tilesN;

  const int wm = (waveId >> 1) & 1;   // wave m-half of macro tile
  const int wn = waveId & 1;          // wave n-half
  const int halfSel = (lane >> 4) & 1;
  const int lrow = lane & 15;

  // loader coordinates: each thread fills 8 consecutive k of one row
  const int lrA = tid >> 2;           // 0..31: A row / B column
  const int lc0 = (tid & 3) << 3;     // 0,8,16,24
  const int gnB = imin(nt * 32 + lrA, N - 1);
#if !GEMM_USE_TDM
  const int gmA = mt * 32 + lrA;
#endif

  v8f c = {};
  for (int kk = 0; kk < K; kk += 32) {
#if GEMM_USE_TDM
    if (waveId == 0) {
      // Tensor DMA descriptor (D#) for the 32x32 f32 A tile; OOB K zero-fills.
      unsigned ldsOff = (unsigned)(uintptr_t)&Asf[0][0];   // low 32b of generic = LDS offset
      unsigned long long ga =
          (unsigned long long)(uintptr_t)(A + (size_t)(mt * 32) * lda + kk);
      unsigned Krem = (unsigned)(K - kk);
      unsigned Mrem = (unsigned)(M - mt * 32);
      u32x4 g0;
      g0[0] = 1u;                                          // count=1 (valid user D#)
      g0[1] = ldsOff;                                      // lds_addr
      g0[2] = (unsigned)(ga & 0xFFFFFFFFu);                // global_addr[31:0]
      g0[3] = (unsigned)((ga >> 32) & 0x1FFFFFFu) | (2u << 30);  // addr[56:32] | type=2
      i32x8 g1;
      g1[0] = (int)(2u << 16);                             // data_size=4B
      g1[1] = (int)((Krem & 0xFFFFu) << 16);               // tensor_dim0[15:0]
      g1[2] = (int)(((Krem >> 16) & 0xFFFFu) | ((Mrem & 0xFFFFu) << 16)); // dim0 hi | dim1 lo
      g1[3] = (int)(((Mrem >> 16) & 0xFFFFu) | (32u << 16)); // dim1 hi | tile_dim0=32
      g1[4] = 32;                                          // tile_dim1=32, tile_dim2=0
      g1[5] = (int)(unsigned)lda;                          // tensor_dim0_stride[31:0]
      g1[6] = 0;
      g1[7] = 0;
      i32x4 gz; gz[0] = 0; gz[1] = 0; gz[2] = 0; gz[3] = 0;
      i32x8 gz8;
      gz8[0] = 0; gz8[1] = 0; gz8[2] = 0; gz8[3] = 0;
      gz8[4] = 0; gz8[5] = 0; gz8[6] = 0; gz8[7] = 0;
      __builtin_amdgcn_tensor_load_to_lds(g0, g1, gz, gz, gz8, 0);
      __builtin_amdgcn_s_wait_tensorcnt(0);
    }
#else
#pragma unroll
    for (int j = 0; j < 8; ++j) {
      int gk = kk + lc0 + j;
      float av = A[(size_t)gmA * lda + imin(gk, K - 1)];
      Asf[lrA][lc0 + j] = (gk < K) ? av : 0.f;
    }
#endif
    // stage B^T tile (branch-free clamp + select), f16
#pragma unroll
    for (int j = 0; j < 8; ++j) {
      int gk = kk + lc0 + j;
      float bv = W[(size_t)imin(gk, K - 1) * N + gnB];
      Bst[lrA][lc0 + j] = (_Float16)(gk < K ? bv : 0.f);
    }
    __syncthreads();

    // fragments per ISA 7.12.2 16-bit layout (two contiguous runs per lane)
    const int arow = wm * 16 + lrow;
    const int bcol = wn * 16 + lrow;
    const float* Ar = &Asf[arow][0];
    v8h b0 = *(const v8h*)&Bst[bcol][halfSel * 8];
    v8h b1 = *(const v8h*)&Bst[bcol][16 + halfSel * 8];
    v16h a, b;
#pragma unroll
    for (int e = 0; e < 8; ++e) {
      a[e]     = (_Float16)Ar[halfSel * 8 + e];
      a[8 + e] = (_Float16)Ar[16 + halfSel * 8 + e];
      b[e] = b0[e]; b[8 + e] = b1[e];
    }
    c = __builtin_amdgcn_wmma_f32_16x16x32_f16(false, a, false, b,
                                               (short)0, c, false, false);
    __syncthreads();
  }

  // epilogue: n fixed per lane, m always in range (M % 32 == 0)
  const int n = nt * 32 + wn * 16 + lrow;
  if (n < N) {
    float bv = bias ? bias[n] : 0.f;
    float* cp = C + (size_t)(mt * 32 + wm * 16 + halfSel * 8) * N + n;
#pragma unroll
    for (int r = 0; r < 8; ++r)
      cp[(size_t)r * N] = apply_act<ACT>(c[r] + bv);
  }
}

// ---------------- LayerNorm (optional residual add, optional relu) ---------
__global__ void ln_kernel(const float* __restrict__ in, const float* __restrict__ res,
                          const float* __restrict__ g, const float* __restrict__ b,
                          float* __restrict__ out, int rows, int D, int relu)
{
  int wave = (int)((blockIdx.x * (size_t)blockDim.x + threadIdx.x) >> 5);
  int lane = threadIdx.x & 31;
  if (wave >= rows) return;
  const float* ip = in + (size_t)wave * D;
  const float* rp = res ? res + (size_t)wave * D : nullptr;
  float s = 0.f, s2 = 0.f;
  for (int i = lane; i < D; i += 32) {
    float v = ip[i] + (rp ? rp[i] : 0.f);
    s += v; s2 += v * v;
  }
#pragma unroll
  for (int o = 16; o > 0; o >>= 1) {
    s  += __shfl_xor(s,  o, 32);
    s2 += __shfl_xor(s2, o, 32);
  }
  float m   = s / (float)D;
  float var = s2 / (float)D - m * m;
  float inv = rsqrtf(var + 1e-5f);
  float* op = out + (size_t)wave * D;
  for (int i = lane; i < D; i += 32) {
    float v = ip[i] + (rp ? rp[i] : 0.f);
    float y = (v - m) * inv * g[i] + b[i];
    if (relu) y = fmaxf(y, 0.f);
    op[i] = y;
  }
}

// ---------------- depthwise causal conv (K=4) + bias + SiLU ----------------
__global__ void conv_silu_kernel(const float* __restrict__ xin, int ldin,
                                 const float* __restrict__ w,
                                 const float* __restrict__ bias,
                                 float* __restrict__ xc,
                                 int nb, int L, int di)
{
  size_t tid = blockIdx.x * (size_t)blockDim.x + threadIdx.x;
  size_t tot = (size_t)nb * L * di;
  if (tid >= tot) return;
  int ch = (int)(tid % di);
  size_t rt = tid / di;
  int t  = (int)(rt % L);
  int bi = (int)(rt / L);
  float acc = bias[ch];
  const float* base = xin + ((size_t)bi * L) * ldin + ch;
#pragma unroll
  for (int k = 0; k < 4; ++k) {
    int tt = t + k - 3;
    if (tt >= 0) acc += base[(size_t)tt * ldin] * w[ch * 4 + k];
  }
  xc[rt * (size_t)di + ch] = silu_f(acc);
}

// ---------------- selective scan (fwd / bwd / inter-slice) -----------------
__global__ void scan_kernel(const float* __restrict__ u, int ldu,
                            const float* __restrict__ delta, int ldd,
                            const float* __restrict__ Bm,
                            const float* __restrict__ Cm, int ldbc,
                            const float* __restrict__ A_log,
                            const float* __restrict__ Dp,
                            float* __restrict__ y, int ldy,
                            int nb, int L, int di, int mode, int addout)
{
  int gid = blockIdx.x * blockDim.x + threadIdx.x;
  if (gid >= nb * di) return;
  int bi = gid / di, ch = gid % di;

  float Ach[16], h[16];
#pragma unroll
  for (int j = 0; j < 16; ++j) {
    Ach[j] = -__expf(A_log[ch * 16 + j]);
    h[j] = 0.f;
  }
  float Dv = Dp[ch];
  int chunk = L / NSLICES;

  for (int t = 0; t < L; ++t) {
    int idx;
    if (mode == 0)      idx = t;
    else if (mode == 1) idx = L - 1 - t;
    else                idx = (t % chunk) * NSLICES + (t / chunk);
    size_t row = (size_t)bi * L + idx;

    if (t + 1 < L) {  // global_prefetch_b8 on next step's operands
      int idn;
      if (mode == 0)      idn = t + 1;
      else if (mode == 1) idn = L - 2 - t;
      else                idn = ((t + 1) % chunk) * NSLICES + ((t + 1) / chunk);
      size_t rn = (size_t)bi * L + idn;
      __builtin_prefetch(&delta[rn * ldd + ch], 0, 0);
      __builtin_prefetch(&Bm[rn * ldbc], 0, 0);
    }

    float dt = delta[row * ldd + ch];
    float uv = u[row * ldu + ch];
    const float* Bt = Bm + row * ldbc;
    const float* Ct = Cm + row * ldbc;
    float dtu = dt * uv;
    float acc = 0.f;
#pragma unroll
    for (int j = 0; j < 16; ++j) {
      h[j] = h[j] * __expf(dt * Ach[j]) + dtu * Bt[j];
      acc += h[j] * Ct[j];
    }
    float outv = acc + uv * Dv;
    float* yp = y + row * ldy + ch;
    *yp = (addout ? *yp : 0.f) + outv;
  }
}

// ---------------- y *= silu(z) ---------------------------------------------
__global__ void gate_silu_kernel(float* __restrict__ y, int ldy,
                                 const float* __restrict__ z, int ldz,
                                 size_t rows, int di)
{
  size_t tid = blockIdx.x * (size_t)blockDim.x + threadIdx.x;
  size_t tot = rows * (size_t)di;
  if (tid >= tot) return;
  size_t r = tid / di; int ch = (int)(tid % di);
  y[r * (size_t)ldy + ch] *= silu_f(z[r * (size_t)ldz + ch]);
}

// ---------------- score head: ind = sigmoid(h @ w + b) ---------------------
__global__ void score_head_kernel(const float* __restrict__ h, int ld,
                                  const float* __restrict__ w,
                                  const float* __restrict__ b,
                                  float* __restrict__ ind, int rows, int K)
{
  int wave = (int)((blockIdx.x * (size_t)blockDim.x + threadIdx.x) >> 5);
  int lane = threadIdx.x & 31;
  if (wave >= rows) return;
  float s = 0.f;
  for (int i = lane; i < K; i += 32) s += h[(size_t)wave * ld + i] * w[i];
#pragma unroll
  for (int o = 16; o > 0; o >>= 1) s += __shfl_xor(s, o, 32);
  if (lane == 0) ind[wave] = 1.f / (1.f + __expf(-(s + b[0])));
}

// ---------------- ad = sigmoid(interp(gse)) + ind --------------------------
__global__ void ad_kernel(const float* __restrict__ gse, int Lg,
                          const float* __restrict__ ind,
                          float* __restrict__ ad, int G, int nb, int S)
{
  size_t tid = blockIdx.x * (size_t)blockDim.x + threadIdx.x;
  size_t tot = (size_t)G * nb * S;
  if (tid >= tot) return;
  int t = (int)(tid % S);
  size_t r = tid / S;
  int bi = (int)(r % nb);
  int g  = (int)(r / nb);
  float pos = (float)t * (float)(Lg - 1) / (float)(S - 1);
  int i0 = (int)floorf(pos);
  if (i0 > Lg - 1) i0 = Lg - 1;
  int i1 = (i0 + 1 < Lg) ? i0 + 1 : Lg - 1;
  float f = pos - (float)i0;
  const float* gr = gse + (size_t)g * Lg;
  float val = gr[i0] + (gr[i1] - gr[i0]) * f;
  float gsv = 1.f / (1.f + __expf(-val));
  ad[tid] = gsv + ind[(size_t)bi * S + t];
}

// ---------------- bitonic argsort per (G,b) row in LDS ---------------------
#define SORT_N 2048
__global__ void sort_kernel(const float* __restrict__ ad,
                            int* __restrict__ idx, int* __restrict__ restore,
                            int S)
{
  __shared__ float key[SORT_N];
  __shared__ int   val[SORT_N];
  int row = blockIdx.x;
  const float* a = ad + (size_t)row * S;
  for (int i = threadIdx.x; i < SORT_N; i += blockDim.x) {
    key[i] = (i < S) ? a[i] : 3.4e38f;
    val[i] = i;
  }
  __syncthreads();
  for (int k = 2; k <= SORT_N; k <<= 1) {
    for (int j = k >> 1; j > 0; j >>= 1) {
      for (int i = threadIdx.x; i < SORT_N; i += blockDim.x) {
        int ixj = i ^ j;
        if (ixj > i) {
          bool up = ((i & k) == 0);
          float ki = key[i], kj = key[ixj];
          if ((ki > kj) == up) {
            key[i] = kj; key[ixj] = ki;
            int tv = val[i]; val[i] = val[ixj]; val[ixj] = tv;
          }
        }
      }
      __syncthreads();
    }
  }
  for (int t = threadIdx.x; t < S; t += blockDim.x) {
    int src = val[t];
    idx[(size_t)row * S + t]       = src;
    restore[(size_t)row * S + src] = t;
  }
}

// ---------------- gather sorted tokens + score channel ---------------------
__global__ void gather_kernel(const float* __restrict__ x,
                              const float* __restrict__ ad,
                              const int* __restrict__ idx,
                              float* __restrict__ out,
                              int G, int nb, int S, int D)
{
  size_t tid = blockIdx.x * (size_t)blockDim.x + threadIdx.x;
  int D1 = D + 1;
  size_t tot = (size_t)G * nb * S * D1;
  if (tid >= tot) return;
  int c = (int)(tid % D1);
  size_t r = tid / D1;
  int t = (int)(r % S);
  size_t gb = r / S;                    // g*nb + bi
  int bi = (int)(gb % nb);
  int src = idx[gb * S + t];
  float v = (c < D) ? x[((size_t)bi * S + src) * D + c] : ad[gb * S + src];
  out[tid] = v;
}

// ---------------- unsort + mean over groups --------------------------------
__global__ void unsort_mean_kernel(const float* __restrict__ y,
                                   const int* __restrict__ restore,
                                   float* __restrict__ out,
                                   int G, int nb, int S, int D1)
{
  size_t tid = blockIdx.x * (size_t)blockDim.x + threadIdx.x;
  size_t tot = (size_t)nb * S * D1;
  if (tid >= tot) return;
  int c = (int)(tid % D1);
  size_t r = tid / D1;
  int t  = (int)(r % S);
  int bi = (int)(r / S);
  float acc = 0.f;
  for (int g = 0; g < G; ++g) {
    size_t gb = (size_t)g * nb + bi;
    int rt = restore[gb * S + t];
    acc += y[(gb * S + rt) * D1 + c];
  }
  out[tid] = acc / (float)G;
}

// ===========================================================================
// Host launch
// ===========================================================================
static inline int cdiv_sz(size_t a, int b) { return (int)((a + (size_t)b - 1) / (size_t)b); }

extern "C" void kernel_launch(void* const* d_in, const int* in_sizes, int n_in,
                              void* d_out, int out_size, void* d_ws, size_t ws_size,
                              hipStream_t stream) {
  (void)in_sizes; (void)n_in; (void)out_size; (void)ws_size;

  // ---- problem constants ----
  const int Bn = 2, S = 2000, DIM = 96;
  const int G = 3, LG = 2048, DEPTHS = 2;
  const int sdi = 192, sdtr = 6;                  // score mamba (d=96)
  const int D1 = 97, ldi = 194, ldtr = 7;         // layer mamba (d=97)
  const int NB2 = G * Bn;                         // 6 group-batched sequences
  const size_t RS = (size_t)Bn * S;               // 4000 rows
  const size_t RL = (size_t)NB2 * S;              // 12000 rows

  // ---- inputs (depth-first dict order) ----
  #define IN(i) ((const float*)d_in[(i)])
  const float* x       = IN(0);
  const float* s_ln_g  = IN(1);
  const float* s_ln_b  = IN(2);
  const float* s_Win   = IN(3);
  const float* s_cw    = IN(4);
  const float* s_cb    = IN(5);
  const float* s_Wx    = IN(6);
  const float* s_Wdt   = IN(7);
  const float* s_bdt   = IN(8);
  const float* s_Alog  = IN(9);
  const float* s_D     = IN(10);
  const float* s_Wout  = IN(11);
  const float* s_Ablog = IN(12);
  const float* s_Aslog = IN(13);
  const float* s_linw  = IN(14);
  const float* s_linb  = IN(15);
  const float* s_gse   = IN(16);
  const float* proj_w  = IN(49);
  const float* proj_b  = IN(50);
  const float* proj_g  = IN(51);
  const float* proj_bb = IN(52);

  // ---- workspace bump allocator ----
  char* wsp = (char*)d_ws;
  auto allocf = [&](size_t n) -> float* {
    float* p = (float*)wsp;
    wsp += ((n * sizeof(float) + 255) / 256) * 256;
    return p;
  };
  auto alloci = [&](size_t n) -> int* {
    int* p = (int*)wsp;
    wsp += ((n * sizeof(int) + 255) / 256) * 256;
    return p;
  };

  float* h      = allocf(RS * DIM);
  float* sxz    = allocf(RS * 2 * sdi);
  float* sxc    = allocf(RS * sdi);
  float* sdxbc  = allocf(RS * (sdtr + 32));
  float* sdelta = allocf(RS * sdi);
  float* sy     = allocf(RS * sdi);
  float* sout   = allocf(RS * DIM);
  float* sind   = allocf(RS);
  float* ad     = allocf((size_t)G * Bn * S);
  int*   idx    = alloci((size_t)G * Bn * S);
  int*   rst    = alloci((size_t)G * Bn * S);
  float* gath   = allocf(RL * D1);
  float* ltmp   = allocf(RL * D1);
  float* lz     = allocf(RL * D1);
  float* lxz    = allocf(RL * 2 * ldi);
  float* lxc    = allocf(RL * ldi);
  float* ldxbc  = allocf(RL * (ldtr + 32));
  float* ldelta = allocf(RL * ldi);
  float* lym    = allocf(RL * ldi);
  float* lmo    = allocf(RL * D1);
  float* ymean  = allocf(RS * D1);
  float* pout   = allocf(RS * DIM);

  auto gemm_grid = [](int M, int N) {
    return ((M + 31) / 32) * ((N + 31) / 32);     // one 32x32 macro-tile per block
  };

  // ================= score path =================
  ln_kernel<<<cdiv_sz(RS * 32, 256), 256, 0, stream>>>(x, nullptr, s_ln_g, s_ln_b, h, (int)RS, DIM, 1);
  wmma_gemm_kernel<0><<<gemm_grid((int)RS, 2 * sdi), 128, 0, stream>>>(h, DIM, s_Win, nullptr, sxz, (int)RS, 2 * sdi, DIM);
  conv_silu_kernel<<<cdiv_sz(RS * sdi, 256), 256, 0, stream>>>(sxz, 2 * sdi, s_cw, s_cb, sxc, Bn, S, sdi);
  wmma_gemm_kernel<0><<<gemm_grid((int)RS, sdtr + 32), 128, 0, stream>>>(sxc, sdi, s_Wx, nullptr, sdxbc, (int)RS, sdtr + 32, sdi);
  wmma_gemm_kernel<2><<<gemm_grid((int)RS, sdi), 128, 0, stream>>>(sdxbc, sdtr + 32, s_Wdt, s_bdt, sdelta, (int)RS, sdi, sdtr);
  {
    int nthr = Bn * sdi;
    int blk = 128, grd = (nthr + blk - 1) / blk;
    scan_kernel<<<grd, blk, 0, stream>>>(sxc, sdi, sdelta, sdi, sdxbc + sdtr, sdxbc + sdtr + 16, sdtr + 32,
                                         s_Alog, s_D, sy, sdi, Bn, S, sdi, 0, 0);
    scan_kernel<<<grd, blk, 0, stream>>>(sxc, sdi, sdelta, sdi, sdxbc + sdtr, sdxbc + sdtr + 16, sdtr + 32,
                                         s_Ablog, s_D, sy, sdi, Bn, S, sdi, 1, 1);
    scan_kernel<<<grd, blk, 0, stream>>>(sxc, sdi, sdelta, sdi, sdxbc + sdtr, sdxbc + sdtr + 16, sdtr + 32,
                                         s_Aslog, s_D, sy, sdi, Bn, S, sdi, 2, 1);
  }
  gate_silu_kernel<<<cdiv_sz(RS * sdi, 256), 256, 0, stream>>>(sy, sdi, sxz + sdi, 2 * sdi, RS, sdi);
  wmma_gemm_kernel<0><<<gemm_grid((int)RS, DIM), 128, 0, stream>>>(sy, sdi, s_Wout, nullptr, sout, (int)RS, DIM, sdi);
  score_head_kernel<<<cdiv_sz(RS * 32, 256), 256, 0, stream>>>(sout, DIM, s_linw, s_linb, sind, (int)RS, DIM);
  ad_kernel<<<cdiv_sz((size_t)G * Bn * S, 256), 256, 0, stream>>>(s_gse, LG, sind, ad, G, Bn, S);
  sort_kernel<<<G * Bn, 1024, 0, stream>>>(ad, idx, rst, S);
  gather_kernel<<<cdiv_sz(RL * D1, 256), 256, 0, stream>>>(x, ad, idx, gath, G, Bn, S, DIM);

  // ================= stacked Mamba layers =================
  for (int l = 0; l < DEPTHS; ++l) {
    int BL = 17 + 16 * l;
    const float* l_linw  = IN(BL + 0);
    const float* l_linb  = IN(BL + 1);
    const float* l_lng   = IN(BL + 2);
    const float* l_lnb   = IN(BL + 3);
    const float* l_Win   = IN(BL + 4);
    const float* l_cw    = IN(BL + 5);
    const float* l_cb    = IN(BL + 6);
    const float* l_Wx    = IN(BL + 7);
    const float* l_Wdt   = IN(BL + 8);
    const float* l_bdt   = IN(BL + 9);
    const float* l_Alog  = IN(BL + 10);
    const float* l_D     = IN(BL + 11);
    const float* l_Wout  = IN(BL + 12);
    const float* l_Ablog = IN(BL + 13);
    const float* l_plng  = IN(BL + 14);
    const float* l_plnb  = IN(BL + 15);

    wmma_gemm_kernel<0><<<gemm_grid((int)RL, D1), 128, 0, stream>>>(gath, D1, l_linw, l_linb, ltmp, (int)RL, D1, D1);
    ln_kernel<<<cdiv_sz(RL * 32, 256), 256, 0, stream>>>(ltmp, nullptr, l_lng, l_lnb, lz, (int)RL, D1, 1);
    wmma_gemm_kernel<0><<<gemm_grid((int)RL, 2 * ldi), 128, 0, stream>>>(lz, D1, l_Win, nullptr, lxz, (int)RL, 2 * ldi, D1);
    conv_silu_kernel<<<cdiv_sz(RL * ldi, 256), 256, 0, stream>>>(lxz, 2 * ldi, l_cw, l_cb, lxc, NB2, S, ldi);
    wmma_gemm_kernel<0><<<gemm_grid((int)RL, ldtr + 32), 128, 0, stream>>>(lxc, ldi, l_Wx, nullptr, ldxbc, (int)RL, ldtr + 32, ldi);
    wmma_gemm_kernel<2><<<gemm_grid((int)RL, ldi), 128, 0, stream>>>(ldxbc, ldtr + 32, l_Wdt, l_bdt, ldelta, (int)RL, ldi, ldtr);
    {
      int nthr = NB2 * ldi;
      int blk = 128, grd = (nthr + blk - 1) / blk;
      scan_kernel<<<grd, blk, 0, stream>>>(lxc, ldi, ldelta, ldi, ldxbc + ldtr, ldxbc + ldtr + 16, ldtr + 32,
                                           l_Alog, l_D, lym, ldi, NB2, S, ldi, 0, 0);
      scan_kernel<<<grd, blk, 0, stream>>>(lxc, ldi, ldelta, ldi, ldxbc + ldtr, ldxbc + ldtr + 16, ldtr + 32,
                                           l_Ablog, l_D, lym, ldi, NB2, S, ldi, 1, 1);
    }
    gate_silu_kernel<<<cdiv_sz(RL * ldi, 256), 256, 0, stream>>>(lym, ldi, lxz + ldi, 2 * ldi, RL, ldi);
    wmma_gemm_kernel<0><<<gemm_grid((int)RL, D1), 128, 0, stream>>>(lym, ldi, l_Wout, nullptr, lmo, (int)RL, D1, ldi);
    ln_kernel<<<cdiv_sz(RL * 32, 256), 256, 0, stream>>>(lmo, gath, l_plng, l_plnb, gath, (int)RL, D1, 0);
  }

  // ================= unsort, mean, projection =================
  unsort_mean_kernel<<<cdiv_sz(RS * D1, 256), 256, 0, stream>>>(gath, rst, ymean, G, Bn, S, D1);
  wmma_gemm_kernel<0><<<gemm_grid((int)RS, DIM), 128, 0, stream>>>(ymean, D1, proj_w, proj_b, pout, (int)RS, DIM, D1);
  ln_kernel<<<cdiv_sz(RS * 32, 256), 256, 0, stream>>>(pout, nullptr, proj_g, proj_bb, (float*)d_out, (int)RS, DIM, 0);

  #undef IN
}
